// RzLinear_21792664060702
// MI455X (gfx1250) — matrix operand order repaired and checked by reading, not compiled, not run on hardware
//
#include <hip/hip_runtime.h>
#include <stdint.h>

// ---------------------------------------------------------------------------
// RzLinear for MI455X (gfx1250, wave32, WMMA)
//   out[M,N] = x[M,K] @ W[K,N] + bias,  W[k,n] = hw[((k*r3+n*r2+r1)%P)%H]
//   M=8192, K=N=4096, H=2^20 (table 4MB -> L2 resident)
// Pipeline:
//   1) rz_premod : rowmod[k]=(k*r3)%P, colmod[n]=(n*r2+r1)%P   (8192 mods)
//   2) rz_build_w: W packed as bf16 k-pairs [k/2][n] in ws (32MB), hash via
//                  32-bit add + conditional subtract + mask (H is pow2)
//   3) rz_gemm   : 128x128x32 tiles, 8 waves, v_wmma_f32_16x16x32_bf16,
//                  double-buffered LDS, register-staged global prefetch held
//                  in named scalars (no arrays/lambdas -> no stack traffic)
// ---------------------------------------------------------------------------

typedef __attribute__((ext_vector_type(16))) __bf16 v16bf;
typedef __attribute__((ext_vector_type(8)))  float  v8f;

#define K_DIM 4096
#define N_DIM 4096
#define BM 128
#define BN 128
#define BK 32
#define LDA 19    // words per A row: 16 bf16-pairs + 3 pad (conflict-free frag loads)
#define LDB 140   // words per B pair-row: 128 + 12 pad (hi-half offset -> +32 banks)

union FragB { v16bf v; uint32_t u[8]; };

__device__ __forceinline__ uint32_t bf16pair(float lo, float hi) {
  uint32_t a = __float_as_uint(lo);
  uint32_t b = __float_as_uint(hi);
  a = (a + 0x7FFFu + ((a >> 16) & 1u)) >> 16;   // round-to-nearest-even
  b = (b + 0x7FFFu + ((b >> 16) & 1u)) >> 16;
  return (b << 16) | (a & 0xFFFFu);
}

// ---- 1) per-row / per-col hash residues (the only 64-bit mods) -------------
__global__ __launch_bounds__(256)
void rz_premod(const long long* __restrict__ rn,
               uint32_t* __restrict__ rowmod, uint32_t* __restrict__ colmod) {
  int i = blockIdx.x * 256 + threadIdx.x;
  uint64_t P  = (uint64_t)rn[0];
  uint64_t r1 = (uint64_t)rn[1];
  uint64_t r2 = (uint64_t)rn[2];
  uint64_t r3 = (uint64_t)rn[3];
  if (i < K_DIM) rowmod[i] = (uint32_t)(((uint64_t)i * r3) % P);
  if (i < N_DIM) colmod[i] = (uint32_t)(((uint64_t)i * r2 + r1) % P);
}

// ---- 2) materialize W as packed bf16 k-pairs: Wp[k/2][n] -------------------
__global__ __launch_bounds__(256)
void rz_build_w(const float* __restrict__ hw, const long long* __restrict__ rn,
                const uint32_t* __restrict__ rowmod,
                const uint32_t* __restrict__ colmod,
                uint32_t hmask, uint32_t* __restrict__ Wp) {
  uint32_t P = (uint32_t)rn[0];                 // P < 2^26
  int gid = blockIdx.x * 256 + threadIdx.x;     // 2M threads total
  int kp  = gid >> 10;                          // 0..2047  (k pair index)
  int n0  = (gid & 1023) << 2;                  // 0..4092
  uint32_t rk0 = rowmod[2 * kp];
  uint32_t rk1 = rowmod[2 * kp + 1];
  uint32_t o[4];
#pragma unroll
  for (int j = 0; j < 4; ++j) {
    uint32_t c  = colmod[n0 + j];
    uint32_t s0 = rk0 + c; if (s0 >= P) s0 -= P;   // (a+b)%P, a,b<P
    uint32_t s1 = rk1 + c; if (s1 >= P) s1 -= P;
    float w0 = hw[s0 & hmask];                     // gather: 4MB table in L2
    float w1 = hw[s1 & hmask];
    o[j] = bf16pair(w0, w1);
  }
  *(uint4*)(Wp + (size_t)kp * N_DIM + n0) = make_uint4(o[0], o[1], o[2], o[3]);
}

// ---- 3) bf16 WMMA GEMM -----------------------------------------------------
// Prefetch tile kt into named registers, then convert+store to LDS.
#define GLOAD(kt)                                                   \
  do {                                                              \
    const uint32_t* bp = bG + (size_t)(kt) * (BK / 2) * N_DIM;      \
    b0 = *(const uint4*)(bp);                                       \
    b1 = *(const uint4*)(bp + 4);                                   \
    const float* ap = aG + (kt) * BK;                               \
    a0 = *(const float4*)(ap);                                      \
    a1 = *(const float4*)(ap + 4);                                  \
    a2 = *(const float4*)(ap + 8);                                  \
    a3 = *(const float4*)(ap + 12);                                 \
  } while (0)

#define STOLDS(buf)                                                 \
  do {                                                              \
    *(uint4*)&sB[buf][bKp][bN0]     = b0;                           \
    *(uint4*)&sB[buf][bKp][bN0 + 4] = b1;                           \
    uint32_t* sa = &sA[buf][aRow][aHalf * 8];                       \
    sa[0] = bf16pair(a0.x, a0.y);                                   \
    sa[1] = bf16pair(a0.z, a0.w);                                   \
    sa[2] = bf16pair(a1.x, a1.y);                                   \
    sa[3] = bf16pair(a1.z, a1.w);                                   \
    sa[4] = bf16pair(a2.x, a2.y);                                   \
    sa[5] = bf16pair(a2.z, a2.w);                                   \
    sa[6] = bf16pair(a3.x, a3.y);                                   \
    sa[7] = bf16pair(a3.z, a3.w);                                   \
  } while (0)

__global__ __launch_bounds__(256) __attribute__((amdgpu_waves_per_eu(2)))
void rz_gemm(const float* __restrict__ x, const uint32_t* __restrict__ Wp,
             const float* __restrict__ bias, float* __restrict__ out) {
  __shared__ __align__(16) uint32_t sA[2][BM][LDA];       // bf16 pairs [m][k/2]
  __shared__ __align__(16) uint32_t sB[2][BK / 2][LDB];   // bf16 pairs [k/2][n]

  const int tid  = threadIdx.x;
  const int lane = tid & 31;
  const int wave = tid >> 5;
  const int lr = lane & 15, hi = lane >> 4;
  const int wm = wave >> 2;                // 0..1 : 64-row patch
  const int wn = wave & 3;                 // 0..3 : 32-col patch
  const int nT = blockIdx.x * BN;
  const int mT = blockIdx.y * BM;

  // staging coordinates
  const int aRow = tid >> 1, aHalf = tid & 1;            // 128 rows x 2 halves
  const int bKp  = tid >> 4, bN0 = (tid & 15) << 3;      // 16 pair-rows x 8 words

  const float*    aG = x  + (size_t)(mT + aRow) * K_DIM + aHalf * 16;
  const uint32_t* bG = Wp + (size_t)bKp * N_DIM + nT + bN0;

  // prefetch staging registers (named scalars -> guaranteed VGPRs)
  float4 a0, a1, a2, a3;
  uint4  b0, b1;

  v8f acc[4][2];
#pragma unroll
  for (int i = 0; i < 4; ++i)
#pragma unroll
    for (int j = 0; j < 2; ++j) acc[i][j] = v8f{};

  const int NK = K_DIM / BK;   // 128
  GLOAD(0);
  STOLDS(0);

  for (int kt = 0; kt < NK; ++kt) {
    __syncthreads();                       // stores of buf visible, prev compute done
    if (kt + 1 < NK) GLOAD(kt + 1);        // hide global latency behind WMMA

    const int buf = kt & 1;
    FragB af[4], bf[2];
#pragma unroll
    for (int fm = 0; fm < 4; ++fm) {       // A 16x32 bf16 frag: kp = v + 8*hi
      int m = wm * 64 + fm * 16 + lr;
#pragma unroll
      for (int v = 0; v < 8; ++v)
        af[fm].u[v] = sA[buf][m][v + hi * 8];
    }
#pragma unroll
    for (int fn = 0; fn < 2; ++fn) {       // B 32x16 bf16 frag: kp = v + 8*hi
      int n = wn * 32 + fn * 16 + lr;
#pragma unroll
      for (int v = 0; v < 8; ++v)
        bf[fn].u[v] = sB[buf][v + hi * 8][n];
    }
#pragma unroll
    for (int fm = 0; fm < 4; ++fm)
#pragma unroll
      for (int fn = 0; fn < 2; ++fn)
        acc[fm][fn] = __builtin_amdgcn_wmma_f32_16x16x32_bf16(
            false, af[fm].v, false, bf[fn].v, (short)0, acc[fm][fn],
            false, false);

    if (kt + 1 < NK) {
      __syncthreads();                     // all waves past compute(kt-1)
      STOLDS((kt + 1) & 1);
    }
  }

  // epilogue: C/D layout -> M = d + 8*hi, N = lane&15
#pragma unroll
  for (int fm = 0; fm < 4; ++fm) {
    int mg = mT + wm * 64 + fm * 16 + hi * 8;
#pragma unroll
    for (int fn = 0; fn < 2; ++fn) {
      int ng = nT + wn * 32 + fn * 16 + lr;
      float bv = bias[ng];
      float* op = out + (size_t)mg * N_DIM + ng;
#pragma unroll
      for (int d = 0; d < 8; ++d)
        op[(size_t)d * N_DIM] = acc[fm][fn][d] + bv;
    }
  }
}

// ---------------------------------------------------------------------------
extern "C" void kernel_launch(void* const* d_in, const int* in_sizes, int n_in,
                              void* d_out, int out_size, void* d_ws, size_t ws_size,
                              hipStream_t stream) {
  const float*     x    = (const float*)d_in[0];
  const float*     hw   = (const float*)d_in[1];
  const long long* rn   = (const long long*)d_in[2];  // int64 [P, r1, r2, r3]
  const float*     bias = (const float*)d_in[3];
  float*           out  = (float*)d_out;

  const int hashSize = in_sizes[1];                   // 2^20
  const uint32_t hmask = (uint32_t)(hashSize - 1);
  const int M = in_sizes[0] / K_DIM;                  // 8192

  uint8_t* ws = (uint8_t*)d_ws;
  uint32_t* rowmod = (uint32_t*)ws;                   // 4096 u32
  uint32_t* colmod = rowmod + K_DIM;                  // 4096 u32
  uint32_t* Wp     = (uint32_t*)(ws + 64 * 1024);     // (K/2)*N u32 = 32MB

  rz_premod<<<16, 256, 0, stream>>>(rn, rowmod, colmod);

  const int wThreads = (K_DIM / 2) * (N_DIM / 4);     // 2M
  rz_build_w<<<wThreads / 256, 256, 0, stream>>>(hw, rn, rowmod, colmod, hmask, Wp);

  dim3 grid(N_DIM / BN, M / BM);                      // 32 x 64
  rz_gemm<<<grid, 256, 0, stream>>>(x, Wp, bias, out);
}
